// GNN_Diffusion_11072425689122
// MI455X (gfx1250) — matrix operand order repaired and checked by reading, not compiled, not run on hardware
//
#include <hip/hip_runtime.h>
#include <hip/hip_bf16.h>

#define NTOT   160000
#define NGENE  5000
#define NEDGE  1200000
#define EPSV   1e-5f

typedef __attribute__((ext_vector_type(16))) _Float16 v16h;
typedef __attribute__((ext_vector_type(8)))  float    v8f;

// ---------------------------------------------------------------- elementwise
__global__ void embed_k(const float* __restrict__ x, const float* __restrict__ emb,
                        const float* __restrict__ pw, const float* __restrict__ pb,
                        const float* __restrict__ gw, const float* __restrict__ gb,
                        float* __restrict__ out, int total) {
  for (int t = blockIdx.x * blockDim.x + threadIdx.x; t < total; t += gridDim.x * blockDim.x) {
    int n = t >> 6, h = t & 63;
    out[t] = x[2 * n] * gw[h] + gb[h] + emb[(n % NGENE) * 64 + h] + x[2 * n + 1] * pw[h] + pb[h];
  }
}

__global__ void deginit_k(float* __restrict__ deg, int n) {
  for (int i = blockIdx.x * blockDim.x + threadIdx.x; i < n; i += gridDim.x * blockDim.x)
    deg[i] = 1.0f;
}

__global__ void degscat_k(const int* __restrict__ dst, const float* __restrict__ ew,
                          float* __restrict__ deg, int e) {
  for (int i = blockIdx.x * blockDim.x + threadIdx.x; i < e; i += gridDim.x * blockDim.x)
    unsafeAtomicAdd(&deg[dst[i]], ew[i]);
}

__global__ void dinv_k(const float* __restrict__ deg, float* __restrict__ dinv, int n) {
  for (int i = blockIdx.x * blockDim.x + threadIdx.x; i < n; i += gridDim.x * blockDim.x)
    dinv[i] = rsqrtf(deg[i]);
}

// out = hw * dinv^2 + bias   (self-loop term + bias of the GCN layer)
__global__ void gcninit_k(const float* __restrict__ hw, const float* __restrict__ dinv,
                          const float* __restrict__ b, float* __restrict__ out, int total) {
  for (int t = blockIdx.x * blockDim.x + threadIdx.x; t < total; t += gridDim.x * blockDim.x) {
    int n = t >> 6, h = t & 63;
    float di = dinv[n];
    out[t] = hw[t] * di * di + b[h];
  }
}

// out[dst] += hw[src] * (dinv[src]*ew*dinv[dst]) — float4 per thread, 16 threads/edge
__global__ void gcnscat_k(const int* __restrict__ src, const int* __restrict__ dst,
                          const float* __restrict__ ew, const float* __restrict__ dinv,
                          const float* __restrict__ hw, float* __restrict__ out, int total) {
  for (int t = blockIdx.x * blockDim.x + threadIdx.x; t < total; t += gridDim.x * blockDim.x) {
    int e = t >> 4, h4 = (t & 15) << 2;
    int s = src[e], d = dst[e];
    float nrm = dinv[s] * ew[e] * dinv[d];
    const float4 v = *(const float4*)(hw + (size_t)s * 64 + h4);
    float* o = out + (size_t)d * 64 + h4;
    unsafeAtomicAdd(o + 0, v.x * nrm);
    unsafeAtomicAdd(o + 1, v.y * nrm);
    unsafeAtomicAdd(o + 2, v.z * nrm);
    unsafeAtomicAdd(o + 3, v.w * nrm);
  }
}

__global__ void relu_k(float* __restrict__ z, int total) {
  for (int i = blockIdx.x * blockDim.x + threadIdx.x; i < total; i += gridDim.x * blockDim.x)
    z[i] = fmaxf(z[i], 0.0f);
}

// ---------------------------------------------------------------- WMMA GEMM
// C[n x MO] = A[n x (KB*32)] @ W[(KB*32) x MO] (+bias). f16 inputs, f32 accum.
// B fragments and bias are hoisted to registers before the strip loop.
template <int KB, int MO, bool BIAS>
__global__ __launch_bounds__(128) void gemm_wmma_k(const float* __restrict__ A,
                                                   const float* __restrict__ W,
                                                   const float* __restrict__ bias,
                                                   float* __restrict__ C, int n) {
  constexpr int K  = KB * 32;
  constexpr int NJ = MO / 16;
  __shared__ _Float16 wlds[K * MO];
  for (int i = threadIdx.x; i < K * MO; i += blockDim.x) wlds[i] = (_Float16)W[i];
  __syncthreads();

  const int lane = threadIdx.x & 31;
  const int m    = lane & 15;   // A row / B col / C col within tile
  const int hi   = lane >> 4;   // half-wave select

  // B fragments: loop-invariant — load once into registers.
  // Layout (32x16): lane holds col m; K = kb*32 + hi*16 + e.
  v16h bfr[NJ * KB];
#pragma unroll
  for (int j = 0; j < NJ; ++j)
#pragma unroll
    for (int kb = 0; kb < KB; ++kb) {
      const _Float16* wp = wlds + (size_t)(kb * 32 + hi * 16) * MO + j * 16 + m;
#pragma unroll
      for (int e = 0; e < 16; ++e) bfr[j * KB + kb][e] = wp[e * MO];
    }
  float bv[NJ];
#pragma unroll
  for (int j = 0; j < NJ; ++j) bv[j] = BIAS ? bias[j * 16 + m] : 0.0f;

  const int nstrips = n >> 4;
  const int wave    = (blockIdx.x * blockDim.x + threadIdx.x) >> 5;
  const int nwaves  = (gridDim.x * blockDim.x) >> 5;

  for (int strip = wave; strip < nstrips; strip += nwaves) {
    const int row0 = strip << 4;
    // prefetch next strip's A row for this lane
    if (strip + nwaves < nstrips)
      __builtin_prefetch(A + (size_t)(row0 + nwaves * 16 + m) * K, 0, 1);
    // A fragments (16x32 f16 per K-block); lanes<16: K={0..15}
    v16h afr[KB];
#pragma unroll
    for (int kb = 0; kb < KB; ++kb) {
      const float* arow = A + (size_t)(row0 + m) * K + kb * 32 + hi * 16;
#pragma unroll
      for (int e = 0; e < 16; ++e)
        afr[kb][e] = (_Float16)arow[e];
    }
#pragma unroll
    for (int j = 0; j < NJ; ++j) {
      v8f acc = {};
#pragma unroll
      for (int kb = 0; kb < KB; ++kb)
        acc = __builtin_amdgcn_wmma_f32_16x16x32_f16(false, afr[kb], false, bfr[j * KB + kb],
                                                     (short)0, acc, false, false);
      const int col = j * 16 + m;
#pragma unroll
      for (int r = 0; r < 8; ++r)    // C/D layout: row = r + 8*hi, col = m
        C[(size_t)(row0 + r + hi * 8) * MO + col] = acc[r] + bv[j];
    }
  }
}

// ---------------------------------------------------------------- batch-norm
__global__ void colstats_k(const float* __restrict__ z, int n, int C,
                           float* __restrict__ mean, float* __restrict__ inv) {
  __shared__ float ssum[256], ssq[256];
  const int j = blockIdx.x;
  float s = 0.0f, q = 0.0f;
  for (int i = threadIdx.x; i < n; i += blockDim.x) {
    float v = z[(size_t)i * C + j];
    s += v; q += v * v;
  }
  ssum[threadIdx.x] = s; ssq[threadIdx.x] = q;
  __syncthreads();
  for (int ofs = 128; ofs > 0; ofs >>= 1) {
    if (threadIdx.x < ofs) {
      ssum[threadIdx.x] += ssum[threadIdx.x + ofs];
      ssq[threadIdx.x]  += ssq[threadIdx.x + ofs];
    }
    __syncthreads();
  }
  if (threadIdx.x == 0) {
    float mu = ssum[0] / (float)n;
    mean[j] = mu;
    inv[j]  = rsqrtf(ssq[0] / (float)n - mu * mu + EPSV);
  }
}

__global__ void bnapply_k(float* __restrict__ z, int total, int C,
                          const float* __restrict__ mean, const float* __restrict__ inv,
                          const float* __restrict__ g, const float* __restrict__ b, int relu) {
  for (int i = blockIdx.x * blockDim.x + threadIdx.x; i < total; i += gridDim.x * blockDim.x) {
    int c = i % C;
    float v = (z[i] - mean[c]) * inv[c] * g[c] + b[c];
    z[i] = relu ? fmaxf(v, 0.0f) : v;
  }
}

// y[n] = dot(h[n,:64], w3) + b3
__global__ void dot64_k(const float* __restrict__ h, const float* __restrict__ w3,
                        const float* __restrict__ b3, float* __restrict__ y, int n) {
  for (int i = blockIdx.x * blockDim.x + threadIdx.x; i < n; i += gridDim.x * blockDim.x) {
    const float* row = h + (size_t)i * 64;
    float s = 0.0f;
#pragma unroll
    for (int k = 0; k < 64; ++k) s += row[k] * w3[k];
    y[i] = s + b3[0];
  }
}

// two-stage deterministic scalar reduction (1024 fixed partials)
__global__ __launch_bounds__(256) void redpart_k(const float* __restrict__ y, int n,
                                                 float* __restrict__ part) {
  __shared__ float ssum[256], ssq[256];
  float s = 0.0f, q = 0.0f;
  for (int i = blockIdx.x * blockDim.x + threadIdx.x; i < n; i += gridDim.x * blockDim.x) {
    float v = y[i]; s += v; q += v * v;
  }
  ssum[threadIdx.x] = s; ssq[threadIdx.x] = q;
  __syncthreads();
  for (int ofs = 128; ofs > 0; ofs >>= 1) {
    if (threadIdx.x < ofs) {
      ssum[threadIdx.x] += ssum[threadIdx.x + ofs];
      ssq[threadIdx.x]  += ssq[threadIdx.x + ofs];
    }
    __syncthreads();
  }
  if (threadIdx.x == 0) { part[blockIdx.x] = ssum[0]; part[1024 + blockIdx.x] = ssq[0]; }
}

__global__ __launch_bounds__(1024) void redfin_k(float* __restrict__ part, int n) {
  __shared__ float ssum[1024], ssq[1024];
  ssum[threadIdx.x] = part[threadIdx.x];
  ssq[threadIdx.x]  = part[1024 + threadIdx.x];
  __syncthreads();
  for (int ofs = 512; ofs > 0; ofs >>= 1) {
    if (threadIdx.x < ofs) {
      ssum[threadIdx.x] += ssum[threadIdx.x + ofs];
      ssq[threadIdx.x]  += ssq[threadIdx.x + ofs];
    }
    __syncthreads();
  }
  if (threadIdx.x == 0) {
    float mu = ssum[0] / (float)n;
    part[2048] = mu;
    part[2049] = rsqrtf(ssq[0] / (float)n - mu * mu + EPSV);
  }
}

__global__ void final_k(const float* __restrict__ y, const float* __restrict__ st,
                        const float* __restrict__ g3, const float* __restrict__ b3,
                        const float* __restrict__ x, float* __restrict__ out, int n) {
  float mu = st[2048], inv = st[2049];
  for (int i = blockIdx.x * blockDim.x + threadIdx.x; i < n; i += gridDim.x * blockDim.x)
    out[i] = (y[i] - mu) * inv * g3[0] + b3[0] + x[2 * i];
}

// ---------------------------------------------------------------- launcher
extern "C" void kernel_launch(void* const* d_in, const int* in_sizes, int n_in,
                              void* d_out, int out_size, void* d_ws, size_t ws_size,
                              hipStream_t stream) {
  (void)in_sizes; (void)n_in; (void)out_size; (void)ws_size;
  const float* x      = (const float*)d_in[0];
  const int*   ei     = (const int*)  d_in[1];
  const float* ew     = (const float*)d_in[2];
  const float* embp   = (const float*)d_in[3];
  const float* pert_w = (const float*)d_in[4];
  const float* pert_b = (const float*)d_in[5];
  const float* gene_w = (const float*)d_in[6];
  const float* gene_b = (const float*)d_in[7];
  const float* gcn_w1 = (const float*)d_in[8];
  const float* gcn_b1 = (const float*)d_in[9];
  const float* gcn_w2 = (const float*)d_in[10];
  const float* gcn_b2 = (const float*)d_in[11];
  const float* mlp_w1 = (const float*)d_in[12];
  const float* mlp_b1 = (const float*)d_in[13];
  const float* bn1_g  = (const float*)d_in[14];
  const float* bn1_b  = (const float*)d_in[15];
  const float* mlp_w2 = (const float*)d_in[16];
  const float* mlp_b2 = (const float*)d_in[17];
  const float* bn2_g  = (const float*)d_in[18];
  const float* bn2_b  = (const float*)d_in[19];
  const float* mlp_w3 = (const float*)d_in[20];
  const float* mlp_b3 = (const float*)d_in[21];
  const float* bn3_g  = (const float*)d_in[22];
  const float* bn3_b  = (const float*)d_in[23];
  const int* src = ei;
  const int* dst = ei + NEDGE;

  // workspace layout (floats)
  float* ws   = (float*)d_ws;
  float* base = ws;                         // N*64
  float* t1   = base + (size_t)NTOT * 64;   // N*64
  float* t2   = t1   + (size_t)NTOT * 64;   // N*64
  float* wide = t2   + (size_t)NTOT * 64;   // N*128
  float* deg  = wide + (size_t)NTOT * 128;  // N
  float* dinv = deg  + NTOT;                // N
  float* yv   = dinv + NTOT;                // N
  float* red  = yv   + NTOT;                // partials + stats
  float* cm   = red + 2304;                 // 128 col means
  float* ci   = cm + 128;                   // 128 col invs

  const int NT64  = NTOT * 64;
  const int NT128 = NTOT * 128;
  const int ESC4  = NEDGE * 16;             // 16 threads/edge, float4 each

  // 1) node embedding
  embed_k<<<4096, 256, 0, stream>>>(x, embp, pert_w, pert_b, gene_w, gene_b, base, NT64);

  // 2) degrees + dinv (shared by both GCN layers)
  deginit_k<<<640, 256, 0, stream>>>(deg, NTOT);
  degscat_k<<<4096, 256, 0, stream>>>(dst, ew, deg, NEDGE);
  dinv_k<<<640, 256, 0, stream>>>(deg, dinv, NTOT);

  // 3) GCN layer 1:  t1 = base@W1 ; t2 = t1*dinv^2 + b1 ; scatter ; relu
  gemm_wmma_k<2, 64, false><<<512, 128, 0, stream>>>(base, gcn_w1, nullptr, t1, NTOT);
  gcninit_k<<<4096, 256, 0, stream>>>(t1, dinv, gcn_b1, t2, NT64);
  gcnscat_k<<<8192, 256, 0, stream>>>(src, dst, ew, dinv, t1, t2, ESC4);
  relu_k<<<4096, 256, 0, stream>>>(t2, NT64);

  // 4) GCN layer 2:  t1 = t2@W2 ; base = t1*dinv^2 + b2 ; scatter (no relu)
  gemm_wmma_k<2, 64, false><<<512, 128, 0, stream>>>(t2, gcn_w2, nullptr, t1, NTOT);
  gcninit_k<<<4096, 256, 0, stream>>>(t1, dinv, gcn_b2, base, NT64);
  gcnscat_k<<<8192, 256, 0, stream>>>(src, dst, ew, dinv, t1, base, ESC4);

  // 5) MLP layer 1: wide = base@mlp_w1 + b1 ; BN ; relu
  gemm_wmma_k<2, 128, true><<<512, 128, 0, stream>>>(base, mlp_w1, mlp_b1, wide, NTOT);
  colstats_k<<<128, 256, 0, stream>>>(wide, NTOT, 128, cm, ci);
  bnapply_k<<<4096, 256, 0, stream>>>(wide, NT128, 128, cm, ci, bn1_g, bn1_b, 1);

  // 6) MLP layer 2: t1 = wide@mlp_w2 + b2 ; BN ; relu
  gemm_wmma_k<4, 64, true><<<512, 128, 0, stream>>>(wide, mlp_w2, mlp_b2, t1, NTOT);
  colstats_k<<<64, 256, 0, stream>>>(t1, NTOT, 64, cm, ci);
  bnapply_k<<<4096, 256, 0, stream>>>(t1, NT64, 64, cm, ci, bn2_g, bn2_b, 1);

  // 7) final column: y = t1@w3 + b3 ; scalar BN ; residual
  dot64_k<<<2048, 256, 0, stream>>>(t1, mlp_w3, mlp_b3, yv, NTOT);
  redpart_k<<<1024, 256, 0, stream>>>(yv, NTOT, red);
  redfin_k<<<1, 1024, 0, stream>>>(red, NTOT);
  final_k<<<640, 256, 0, stream>>>(yv, red, bn3_g, bn3_b, x, (float*)d_out, NTOT);
}
